// MultiHeadAttention_23931557773611
// MI455X (gfx1250) — compile-verified, matrix operand order checked by reading, not run on hardware
//
#include <hip/hip_runtime.h>

// Problem constants
#define B_   4
#define T_   2048
#define DIM_ 1024
#define NH_  16
#define HD_  64
#define BT_  (B_ * T_)

typedef __attribute__((ext_vector_type(16))) __bf16 v16bf;
typedef __attribute__((ext_vector_type(8)))  float  v8f;
typedef __attribute__((ext_vector_type(4)))  unsigned u32x4;
typedef __attribute__((ext_vector_type(8)))  unsigned u32x8;

// ---------------------------------------------------------------------------
// helpers
// ---------------------------------------------------------------------------
static __device__ __forceinline__ unsigned short f32_bf16(float f) {
    union { float f; unsigned u; } v;
    v.f = f;
    unsigned r = v.u + 0x7FFFu + ((v.u >> 16) & 1u); // round-to-nearest-even
    return (unsigned short)(r >> 16);
}

// Load one 16-bit A/B WMMA fragment whose per-lane data is K-contiguous:
// elements j=0..7 at p[0..7], j=8..15 at p[16..23] (ISA 7.12.2 layout with
// the lane-half offset h*8 already folded into p by the caller).
struct U32x8s { uint4 a, b; };
static __device__ __forceinline__ v16bf load_frag(const unsigned short* p) {
    U32x8s t;
    t.a = *(const uint4*)p;
    t.b = *(const uint4*)(p + 16);
    return __builtin_bit_cast(v16bf, t);
}

// 16-lane row reductions via DPP row rotations (pure VALU, no LDS traffic).
static __device__ __forceinline__ float red16_max(float x) {
    int t;
    t = __builtin_amdgcn_update_dpp(0, __builtin_bit_cast(int, x), 0x121, 0xF, 0xF, true);
    x = fmaxf(x, __builtin_bit_cast(float, t));
    t = __builtin_amdgcn_update_dpp(0, __builtin_bit_cast(int, x), 0x122, 0xF, 0xF, true);
    x = fmaxf(x, __builtin_bit_cast(float, t));
    t = __builtin_amdgcn_update_dpp(0, __builtin_bit_cast(int, x), 0x124, 0xF, 0xF, true);
    x = fmaxf(x, __builtin_bit_cast(float, t));
    t = __builtin_amdgcn_update_dpp(0, __builtin_bit_cast(int, x), 0x128, 0xF, 0xF, true);
    x = fmaxf(x, __builtin_bit_cast(float, t));
    return x;
}
static __device__ __forceinline__ float red16_sum(float x) {
    int t;
    t = __builtin_amdgcn_update_dpp(0, __builtin_bit_cast(int, x), 0x121, 0xF, 0xF, true);
    x += __builtin_bit_cast(float, t);
    t = __builtin_amdgcn_update_dpp(0, __builtin_bit_cast(int, x), 0x122, 0xF, 0xF, true);
    x += __builtin_bit_cast(float, t);
    t = __builtin_amdgcn_update_dpp(0, __builtin_bit_cast(int, x), 0x124, 0xF, 0xF, true);
    x += __builtin_bit_cast(float, t);
    t = __builtin_amdgcn_update_dpp(0, __builtin_bit_cast(int, x), 0x128, 0xF, 0xF, true);
    x += __builtin_bit_cast(float, t);
    return x;
}

// ---------------------------------------------------------------------------
// TDM: issue a 2D bf16 tile load global->LDS (CDNA5 ISA ch.8, D# groups 0/1).
// lds_off   : byte offset of destination in LDS (tile stored row-major packed)
// gaddr     : global byte address of tile start
// tensor_d0 : tensor dim-0 length (elements)   tensor_d1: dim-1 length (rows)
// d0_stride : elements between consecutive rows
// tile_d0   : tile row length (elements)       tile_d1 : tile row count
// ---------------------------------------------------------------------------
static __device__ __forceinline__ void tdm_load_2d_bf16(
    unsigned lds_off, unsigned long long gaddr,
    unsigned tensor_d0, unsigned tensor_d1,
    unsigned long long d0_stride,
    unsigned tile_d0, unsigned tile_d1) {
    u32x4 g0;
    g0[0] = 1u;                                   // count=1 (valid), user mode
    g0[1] = lds_off;                              // lds_addr (bytes)
    g0[2] = (unsigned)gaddr;                      // global_addr[31:0]
    g0[3] = ((unsigned)(gaddr >> 32) & 0x01FFFFFFu) | 0x80000000u; // [56:32] | type=2
    u32x8 g1;
    g1[0] = 1u << 16;                             // data_size=1 (2 bytes), mask=0
    g1[1] = (tensor_d0 & 0xFFFFu) << 16;          // tensor_dim0[15:0]
    g1[2] = (tensor_d0 >> 16) | ((tensor_d1 & 0xFFFFu) << 16); // d0[31:16] | d1[15:0]
    g1[3] = (tensor_d1 >> 16) | (tile_d0 << 16);  // d1[31:16] | tile_dim0
    g1[4] = tile_d1;                              // tile_dim1, tile_dim2=0
    g1[5] = (unsigned)d0_stride;                  // tensor_dim0_stride[31:0]
    g1[6] = (unsigned)(d0_stride >> 32) & 0xFFFFu;// stride[47:32], dim1_stride=0
    g1[7] = 0u;
    asm volatile("tensor_load_to_lds %0, %1" :: "s"(g0), "s"(g1) : "memory");
}

// ---------------------------------------------------------------------------
// fp32 -> bf16 conversion (plain)
// ---------------------------------------------------------------------------
__global__ void mha_cvt_bf16(const float* __restrict__ in,
                             unsigned short* __restrict__ out, int n) {
    int i = blockIdx.x * blockDim.x + threadIdx.x;
    if (i < n) out[i] = f32_bf16(in[i]);
}

// ---------------------------------------------------------------------------
// fp32 [K,N] -> bf16 transposed [N,K] (tiled through LDS, both sides coalesced)
// ---------------------------------------------------------------------------
__global__ __launch_bounds__(256)
void mha_cvt_bf16_t(const float* __restrict__ in,
                    unsigned short* __restrict__ out) {
    __shared__ unsigned short tile[64][65];
    const int k0 = blockIdx.y * 64, n0 = blockIdx.x * 64;
    const int tid = threadIdx.x;
    #pragma unroll
    for (int i = 0; i < 16; i++) {
        int idx = i * 256 + tid;            // 0..4095
        int rr = idx >> 6, cc = idx & 63;
        tile[rr][cc] = f32_bf16(in[(size_t)(k0 + rr) * DIM_ + n0 + cc]);
    }
    __syncthreads();
    #pragma unroll
    for (int i = 0; i < 16; i++) {
        int idx = i * 256 + tid;
        int rr = idx >> 6, cc = idx & 63;   // rr: local n, cc: local k
        out[(size_t)(n0 + rr) * DIM_ + k0 + cc] = tile[cc][rr];
    }
}

// ---------------------------------------------------------------------------
// bf16 GEMM:  C[M,N] = A[M,K] @ Bt[N,K]^T + bias[N]
// Block tile 128x128, BLK_K=64, 8 waves, wave tile 32x64.
// A/B tiles staged by the Tensor Data Mover, double-buffered so the DMA for
// chunk i+1 overlaps the WMMAs of chunk i (TENSORcnt <= 1 pipelining).
// MODE 1: bf16 scatter to [B,NH,T,HD]   (Q, K)
// MODE 3: bf16 scatter to [B,NH,HD,T]   (V, pre-transposed for attention)
// MODE 2: fp32 row-major out            (final projection)
// ---------------------------------------------------------------------------
template <int MODE>
__global__ __launch_bounds__(256)
void mha_gemm_bf16(const unsigned short* __restrict__ A,
                   const unsigned short* __restrict__ Bt,
                   const float* __restrict__ bias,
                   void* __restrict__ outp,
                   int M, int N, int K) {
    __shared__ __align__(16) unsigned short As[2][128 * 64];
    __shared__ __align__(16) unsigned short Bs[2][128 * 64];

    const int tid  = threadIdx.x;
    const int lane = tid & 31;
    const int w    = tid >> 5;     // 0..7
    const int wm   = w >> 1;       // 0..3 : 32-row slab
    const int wn   = w & 1;        // 0..1 : 64-col slab
    const int m0   = blockIdx.y * 128;
    const int n0   = blockIdx.x * 128;
    const int r    = lane & 15;
    const int h    = lane >> 4;

    const unsigned asbase = (unsigned)(unsigned long long)(void*)&As[0][0];
    const unsigned bsbase = (unsigned)(unsigned long long)(void*)&Bs[0][0];
    const unsigned long long agbase = (unsigned long long)(size_t)A + (size_t)m0 * K * 2;
    const unsigned long long bgbase = (unsigned long long)(size_t)Bt + (size_t)n0 * K * 2;

    v8f acc[2][4];
    #pragma unroll
    for (int a = 0; a < 2; a++)
        #pragma unroll
        for (int b = 0; b < 4; b++) {
            v8f z = {0.f, 0.f, 0.f, 0.f, 0.f, 0.f, 0.f, 0.f};
            acc[a][b] = z;
        }

    const int nk = K / 64;
    // prologue: DMA chunk 0 into buffer 0
    if (w == 0)      tdm_load_2d_bf16(asbase, agbase, K, M, K, 64, 128);
    else if (w == 1) tdm_load_2d_bf16(bsbase, bgbase, K, N, K, 64, 128);

    for (int i = 0; i < nk; i++) {
        const int p = i & 1;
        __syncthreads();                       // back buffer free (prev reads done)
        if (i + 1 < nk) {                      // DMA chunk i+1 into back buffer
            unsigned long long gk = (unsigned long long)(i + 1) * 64 * 2;
            if (w == 0)
                tdm_load_2d_bf16(asbase + (unsigned)(1 - p) * 16384,
                                 agbase + gk, K, M, K, 64, 128);
            else if (w == 1)
                tdm_load_2d_bf16(bsbase + (unsigned)(1 - p) * 16384,
                                 bgbase + gk, K, N, K, 64, 128);
            if (w <= 1) __builtin_amdgcn_s_wait_tensorcnt(1); // chunk i complete
        } else {
            if (w <= 1) __builtin_amdgcn_s_wait_tensorcnt(0);
        }
        __syncthreads();                       // chunk i published

        #pragma unroll
        for (int ks = 0; ks < 64; ks += 32) {
            v16bf af[2], bfv[4];
            #pragma unroll
            for (int mt = 0; mt < 2; mt++)
                af[mt] = load_frag(&As[p][(wm * 32 + mt * 16 + r) * 64 + ks + h * 8]);
            #pragma unroll
            for (int nt = 0; nt < 4; nt++)
                bfv[nt] = load_frag(&Bs[p][(wn * 64 + nt * 16 + r) * 64 + ks + h * 8]);

            #pragma unroll
            for (int mt = 0; mt < 2; mt++)
                #pragma unroll
                for (int nt = 0; nt < 4; nt++)
                    acc[mt][nt] = __builtin_amdgcn_wmma_f32_16x16x32_bf16(
                        false, af[mt], false, bfv[nt], (short)0, acc[mt][nt],
                        false, false);
        }
    }

    // epilogue
    #pragma unroll
    for (int mt = 0; mt < 2; mt++) {
        #pragma unroll
        for (int nt = 0; nt < 4; nt++) {
            #pragma unroll
            for (int j = 0; j < 8; j++) {
                int m = m0 + wm * 32 + mt * 16 + j + 8 * h;
                int n = n0 + wn * 64 + nt * 16 + r;
                float val = acc[mt][nt][j] + bias[n];
                if (MODE == 1) {
                    int b  = m / T_, t = m % T_;
                    int hh = n / HD_, d = n % HD_;
                    ((unsigned short*)outp)
                        [(((size_t)b * NH_ + hh) * T_ + t) * HD_ + d] = f32_bf16(val);
                } else if (MODE == 3) {
                    int b  = m / T_, t = m % T_;
                    int hh = n / HD_, d = n % HD_;
                    ((unsigned short*)outp)
                        [(((size_t)b * NH_ + hh) * HD_ + d) * T_ + t] = f32_bf16(val);
                } else {
                    ((float*)outp)[(size_t)m * N + n] = val;
                }
            }
        }
    }
}

// ---------------------------------------------------------------------------
// Flash attention. Q/K bf16 [B,NH,T,HD]; V bf16 [B,NH,HD,T] (pre-transposed).
// Out: bf16 [B,T,DIM]. grid = (B*NH, T/128); 8 waves; wave owns 16 q rows.
// K/V tiles staged by TDM, double-buffered (DMA overlaps WMMA + softmax).
// ---------------------------------------------------------------------------
__global__ __launch_bounds__(256)
void mha_flash_attn(const unsigned short* __restrict__ Q,
                    const unsigned short* __restrict__ Kt,
                    const unsigned short* __restrict__ Vt,
                    unsigned short* __restrict__ O) {
    __shared__ __align__(16) unsigned short Ks[2][32 * 64];   // [key][dim]
    __shared__ __align__(16) unsigned short Vs[2][64 * 32];   // [dim][key]
    __shared__ __align__(16) unsigned short Pscr[8][16 * 32]; // [row][key]

    const int tid  = threadIdx.x;
    const int lane = tid & 31;
    const int w    = tid >> 5;
    const int bh   = blockIdx.x;        // b*NH + head
    const int head = bh & (NH_ - 1);
    const int bidx = bh >> 4;           // NH_ == 16
    const int qbase = blockIdx.y * 128 + w * 16;
    const int r = lane & 15;
    const int h = lane >> 4;

    const unsigned short* Qh = Q  + (size_t)bh * T_ * HD_;
    const unsigned short* Kh = Kt + (size_t)bh * T_ * HD_;
    const unsigned short* Vh = Vt + (size_t)bh * HD_ * T_;

    const unsigned ksbase = (unsigned)(unsigned long long)(void*)&Ks[0][0];
    const unsigned vsbase = (unsigned)(unsigned long long)(void*)&Vs[0][0];
    const unsigned long long kgbase = (unsigned long long)(size_t)Kh;
    const unsigned long long vgbase = (unsigned long long)(size_t)Vh;

    // wave's 16x64 Q tile as two A fragments (dims 0..31 / 32..63)
    v16bf qf[2];
    #pragma unroll
    for (int half = 0; half < 2; half++)
        qf[half] = load_frag(&Qh[(size_t)(qbase + r) * HD_ + half * 32 + h * 8]);

    v8f o[4];
    #pragma unroll
    for (int f = 0; f < 4; f++) {
        v8f z = {0.f, 0.f, 0.f, 0.f, 0.f, 0.f, 0.f, 0.f};
        o[f] = z;
    }
    float rm[8], rs[8];
    #pragma unroll
    for (int j = 0; j < 8; j++) { rm[j] = -3.0e38f; rs[j] = 0.f; }

    const float scale = 0.125f; // HD^-0.5
    const int nkc = T_ / 32;

    // prologue: DMA chunk 0 into buffer 0
    if (w == 0)      tdm_load_2d_bf16(ksbase, kgbase, HD_, T_, HD_, HD_, 32);
    else if (w == 1) tdm_load_2d_bf16(vsbase, vgbase, T_, HD_, T_, 32, HD_);

    for (int kc = 0; kc < nkc; kc++) {
        const int p = kc & 1;
        __syncthreads();                       // back buffer free
        if (kc + 1 < nkc) {                    // DMA chunk kc+1 into back buffer
            if (w == 0)
                tdm_load_2d_bf16(ksbase + (unsigned)(1 - p) * 4096,
                    kgbase + (unsigned long long)(kc + 1) * 32 * HD_ * 2,
                    HD_, T_, HD_, HD_, 32);
            else if (w == 1)
                tdm_load_2d_bf16(vsbase + (unsigned)(1 - p) * 4096,
                    vgbase + (unsigned long long)(kc + 1) * 32 * 2,
                    T_, HD_, T_, 32, HD_);
            if (w <= 1) __builtin_amdgcn_s_wait_tensorcnt(1);
        } else {
            if (w <= 1) __builtin_amdgcn_s_wait_tensorcnt(0);
        }
        __syncthreads();                       // chunk kc published

        // S[16 x 32] as two 16x16 tiles: per tile two WMMAs over HD=64
        v8f s[2];
        #pragma unroll
        for (int st = 0; st < 2; st++) {
            v16bf kf0 = load_frag(&Ks[p][(st * 16 + r) * 64 + h * 8]);       // dims 0..31
            v16bf kf1 = load_frag(&Ks[p][(st * 16 + r) * 64 + 32 + h * 8]);  // dims 32..63
            v8f z = {0.f, 0.f, 0.f, 0.f, 0.f, 0.f, 0.f, 0.f};
            z = __builtin_amdgcn_wmma_f32_16x16x32_bf16(
                false, qf[0], false, kf0, (short)0, z, false, false);
            z = __builtin_amdgcn_wmma_f32_16x16x32_bf16(
                false, qf[1], false, kf1, (short)0, z, false, false);
            s[st] = z;
        }

        // online softmax per row (row j+8*h lives in VGPR j of this lane half)
        #pragma unroll
        for (int j = 0; j < 8; j++) {
            float s0 = s[0][j] * scale;
            float s1 = s[1][j] * scale;
            float mx   = red16_max(fmaxf(s0, s1));
            float newm = fmaxf(rm[j], mx);
            float alpha = __expf(rm[j] - newm);
            float p0 = __expf(s0 - newm);
            float p1 = __expf(s1 - newm);
            rs[j] = rs[j] * alpha + red16_sum(p0 + p1);
            rm[j] = newm;
            #pragma unroll
            for (int f = 0; f < 4; f++) o[f][j] *= alpha;
            int row = j + 8 * h;
            Pscr[w][row * 32 + r]      = f32_bf16(p0);
            Pscr[w][row * 32 + 16 + r] = f32_bf16(p1);
        }
        __syncthreads();

        // O += P[16x32] @ V[32x64] : 4 WMMAs over the 4 dim-tiles
        v16bf pf = load_frag(&Pscr[w][r * 32 + h * 8]);
        #pragma unroll
        for (int f = 0; f < 4; f++) {
            v16bf vf = load_frag(&Vs[p][(f * 16 + r) * 32 + h * 8]);
            o[f] = __builtin_amdgcn_wmma_f32_16x16x32_bf16(
                false, pf, false, vf, (short)0, o[f], false, false);
        }
    }

    // normalize + write to [B, T, DIM] bf16 (heads re-interleaved)
    #pragma unroll
    for (int j = 0; j < 8; j++) {
        float inv = 1.0f / rs[j];
        int q = qbase + j + 8 * h;
        #pragma unroll
        for (int f = 0; f < 4; f++) {
            O[((size_t)bidx * T_ + q) * DIM_ + head * HD_ + f * 16 + r] =
                f32_bf16(o[f][j] * inv);
        }
    }
}

// ---------------------------------------------------------------------------
// launcher
// ---------------------------------------------------------------------------
extern "C" void kernel_launch(void* const* d_in, const int* in_sizes, int n_in,
                              void* d_out, int out_size, void* d_ws,
                              size_t ws_size, hipStream_t stream) {
    const float* x  = (const float*)d_in[0];
    const float* Wq = (const float*)d_in[1];
    const float* bq = (const float*)d_in[2];
    const float* Wk = (const float*)d_in[3];
    const float* bk = (const float*)d_in[4];
    const float* Wv = (const float*)d_in[5];
    const float* bv = (const float*)d_in[6];
    const float* Wo = (const float*)d_in[7];
    const float* bo = (const float*)d_in[8];

    const size_t NX = (size_t)BT_ * DIM_;   // 8388608
    const size_t NW = (size_t)DIM_ * DIM_;  // 1048576

    unsigned short* xb  = (unsigned short*)d_ws;
    unsigned short* wqt = xb + NX;          // W^T [N,K] bf16
    unsigned short* wkt = wqt + NW;
    unsigned short* wvt = wkt + NW;
    unsigned short* wot = wvt + NW;
    unsigned short* Qb  = wot + NW;         // [B,NH,T,HD]
    unsigned short* Kb  = Qb + NX;          // [B,NH,T,HD]
    unsigned short* Vb  = Kb + NX;          // [B,NH,HD,T]  (transposed)
    unsigned short* Ab  = Vb + NX;          // [B,T,DIM]

    // conversions
    mha_cvt_bf16<<<((int)NX + 255) / 256, 256, 0, stream>>>(x, xb, (int)NX);
    dim3 gt(DIM_ / 64, DIM_ / 64);
    mha_cvt_bf16_t<<<gt, 256, 0, stream>>>(Wq, wqt);
    mha_cvt_bf16_t<<<gt, 256, 0, stream>>>(Wk, wkt);
    mha_cvt_bf16_t<<<gt, 256, 0, stream>>>(Wv, wvt);
    mha_cvt_bf16_t<<<gt, 256, 0, stream>>>(Wo, wot);

    // QKV projections (TDM-staged, double-buffered)
    dim3 gg(DIM_ / 128, BT_ / 128);
    mha_gemm_bf16<1><<<gg, 256, 0, stream>>>(xb, wqt, bq, Qb, BT_, DIM_, DIM_);
    mha_gemm_bf16<1><<<gg, 256, 0, stream>>>(xb, wkt, bk, Kb, BT_, DIM_, DIM_);
    mha_gemm_bf16<3><<<gg, 256, 0, stream>>>(xb, wvt, bv, Vb, BT_, DIM_, DIM_);

    // flash attention (TDM-staged K/V, double-buffered)
    mha_flash_attn<<<dim3(B_ * NH_, T_ / 128), 256, 0, stream>>>(Qb, Kb, Vb, Ab);

    // output projection -> fp32 d_out
    mha_gemm_bf16<2><<<gg, 256, 0, stream>>>(Ab, wot, bo, (float*)d_out,
                                             BT_, DIM_, DIM_);
}
